// MOORENetwork_21320217658194
// MI455X (gfx1250) — compile-verified, matrix operand order checked by reading, not run on hardware
//
#include <hip/hip_runtime.h>
#include <hip/hip_bf16.h>

// ---------------------------------------------------------------------------
// MI455X (gfx1250). All matrix math on bf16 WMMA (v_wmma_f32_16x16x32_bf16),
// fp32 accumulate. Weights pre-transposed once to [.][N][K] so A and B tiles
// are straight row copies -> GLOBAL_LOAD_ASYNC_TO_LDS_B128 double-buffered
// staging. Wave tile 16Mx64N (acc[4], one A fragment reused over 4 B
// fragments): this shape was verified to produce mov/NOP-free WMMA codegen.
// The per-task head is tid-bucketed and also runs on WMMA.
// ---------------------------------------------------------------------------

#if __has_builtin(__builtin_amdgcn_global_load_async_to_lds_b128)
#define ASYNC_LDS 1
#else
#define ASYNC_LDS 0
#endif

typedef __attribute__((__vector_size__(4 * sizeof(int)))) int v4i;

#if ASYNC_LDS
#if __has_builtin(__builtin_amdgcn_s_wait_asynccnt)
#define WAIT_ASYNC() __builtin_amdgcn_s_wait_asynccnt(0)
#else
#define WAIT_ASYNC() asm volatile("s_wait_asynccnt 0x0" ::: "memory")
#endif
__device__ __forceinline__ void async_copy_b128(const void* g, void* l) {
    __builtin_amdgcn_global_load_async_to_lds_b128(
        (__attribute__((address_space(1))) v4i*)g,
        (__attribute__((address_space(3))) v4i*)l, 0, 0);
}
#endif

typedef __attribute__((ext_vector_type(16))) __bf16 v16bf;
typedef __attribute__((ext_vector_type(8)))  float  v8f;

static constexpr int B_    = 4096;
static constexpr int FEAT_ = 512;
static constexpr int WID_  = 1024;
static constexpr int E_    = 8;
static constexpr int T_    = 50;
static constexpr int HD_   = 256;
static constexpr int XD_   = FEAT_ + T_;   // 562

// tiling (register-pressure-verified shape)
static constexpr int MT  = 64;    // block M
static constexpr int NT  = 128;   // block N
static constexpr int KT  = 32;    // K slice (bf16 WMMA depth)
static constexpr int LSTR = 40;   // LDS row stride in elements (80B)

// ---------------- fp32 [Z][K][N] -> bf16 [Z][N][K] (32x32 LDS transpose) ----
__global__ __launch_bounds__(256)
void transpose_to_bf16_kernel(const float* __restrict__ src,
                              __bf16* __restrict__ dst, int K, int N) {
    __shared__ float tile[32][33];
    const int z  = blockIdx.z;
    const int k0 = blockIdx.x * 32;
    const int n0 = blockIdx.y * 32;
    const int tx = threadIdx.x & 31;
    const int ty = threadIdx.x >> 5;
#pragma unroll
    for (int r = 0; r < 32; r += 8)
        tile[ty + r][tx] = src[((size_t)z * K + k0 + ty + r) * N + n0 + tx];
    __syncthreads();
#pragma unroll
    for (int r = 0; r < 32; r += 8)
        dst[((size_t)z * N + n0 + ty + r) * K + k0 + tx] = (__bf16)tile[tx][ty + r];
}

// ---------------- strip one-hot columns, convert feats to bf16 --------------
__global__ void extract_x_kernel(const float* __restrict__ x,
                                 __bf16* __restrict__ xb) {
    size_t i = (size_t)blockIdx.x * blockDim.x + threadIdx.x;  // B*FEAT total
    size_t b = i >> 9;
    size_t f = i & 511;
    xb[i] = (__bf16)x[b * XD_ + f];
}

// ---------------- tid = argmax(onehot), emb = W_emb[tid] + b_emb ------------
__global__ void emb_tid_kernel(const float* __restrict__ x,
                               const float* __restrict__ W_emb,
                               const float* __restrict__ b_emb,
                               float* __restrict__ emb,
                               int* __restrict__ tid) {
    int b = blockIdx.x * blockDim.x + threadIdx.x;
    if (b >= B_) return;
    const float* oh = x + (size_t)b * XD_ + FEAT_;
    int   ti   = 0;
    float best = oh[0];
    for (int i = 1; i < T_; ++i) {
        float v = oh[i];
        if (v > best) { best = v; ti = i; }
    }
    tid[b] = ti;
    for (int e = 0; e < E_; ++e)
        emb[(size_t)b * E_ + e] = W_emb[(size_t)ti * E_ + e] + b_emb[e];
}

// ---------------- tid bucketing (histogram / scan / scatter) ----------------
__global__ void bucket_init_kernel(int* cnt, int* cur) {
    int i = threadIdx.x;
    if (i < T_) { cnt[i] = 0; cur[i] = 0; }
}
__global__ void bucket_count_kernel(const int* __restrict__ tid, int* cnt) {
    int b = blockIdx.x * 256 + threadIdx.x;
    if (b < B_) atomicAdd(&cnt[tid[b]], 1);
}
__global__ void bucket_scan_kernel(const int* __restrict__ cnt, int* off) {
    if (threadIdx.x == 0) {
        int s = 0;
        for (int i = 0; i < T_; ++i) { off[i] = s; s += cnt[i]; }
        off[T_] = s;
    }
}
__global__ void bucket_scatter_kernel(const int* __restrict__ tid,
                                      const int* __restrict__ off,
                                      int* cur, int* perm) {
    int b = blockIdx.x * 256 + threadIdx.x;
    if (b >= B_) return;
    int tk = tid[b];
    int p  = off[tk] + atomicAdd(&cur[tk], 1);
    perm[p] = b;
}

// ---------------- batched-expert bf16 WMMA GEMM -----------------------------
// out[b, e, n] = act( in[b,(e,)k] @ W[e,k,n] + bias[e,n] ); Wt is [E][N][K].
// Block 64M x 128N; 8 waves as 4(M) x 2(N); each wave 16x64 -> acc[4],
// one A fragment reused over 4 B fragments. K staged 32-deep, double-buffered.
template <bool IN_PER_EXPERT, bool RELU, bool OUT_BF16>
__global__ __launch_bounds__(256)
void gemm_moe_kernel(const __bf16* __restrict__ A,
                     const __bf16* __restrict__ Wt,   // [E][N][K]
                     const float*  __restrict__ bias,
                     void* __restrict__ outv,
                     int K, int N) {
    __shared__ __align__(16) __bf16 lA[ASYNC_LDS ? 2 : 1][MT * LSTR];
    __shared__ __align__(16) __bf16 lB[ASYNC_LDS ? 2 : 1][NT * LSTR];

    const int t     = threadIdx.x;
    const int e     = blockIdx.z;
    const size_t b0 = (size_t)blockIdx.x * MT;
    const int n0    = blockIdx.y * NT;

    const int lane = t & 31;
    const int wave = t >> 5;
    const int mrow = (wave & 3) * 16;     // wave M offset
    const int ncol = (wave >> 2) * 64;    // wave N offset (4 x 16 cols)
    const int ln   = lane & 15;
    const int hi   = lane >> 4;           // K-half select per ISA 7.12.2

    v8f acc[4] = {};

    // cooperative-load coordinates (all 16B-chunk copies)
    const int ar = t >> 2, ac = (t & 3) * 8;    // A: 64 rows x 4 chunks
    const int bn = t >> 1, bc = (t & 1) * 16;   // B: 128 rows x 2x(2 chunks)

    const __bf16* Ap = A +
        (IN_PER_EXPERT ? ((b0 + ar) * (size_t)E_ + e) * (size_t)K
                       : (b0 + ar) * (size_t)K) + ac;
    const __bf16* Bp = Wt + ((size_t)e * N + n0 + bn) * (size_t)K + bc;

    const int nK = K / KT;

    auto compute = [&](int buf) {
        union F { uint4 q[2]; v16bf v; };
        F af;
        const __bf16* la = &lA[buf][(mrow + ln) * LSTR + hi * 8];
        af.q[0] = *(const uint4*)(la);
        af.q[1] = *(const uint4*)(la + 16);
#pragma unroll
        for (int j = 0; j < 4; ++j) {
            F bf;
            const __bf16* lb = &lB[buf][(ncol + j * 16 + ln) * LSTR + hi * 8];
            bf.q[0] = *(const uint4*)(lb);
            bf.q[1] = *(const uint4*)(lb + 16);
            acc[j] = __builtin_amdgcn_wmma_f32_16x16x32_bf16(
                false, af.v, false, bf.v, (short)0, acc[j], false, false);
        }
    };

#if ASYNC_LDS
    auto stage = [&](int buf, int k0) {   // 3 async b128 copies per thread
        async_copy_b128(Ap + k0, &lA[buf][ar * LSTR + ac]);
        async_copy_b128(Bp + k0,     &lB[buf][bn * LSTR + bc]);
        async_copy_b128(Bp + k0 + 8, &lB[buf][bn * LSTR + bc + 8]);
    };
    stage(0, 0);
    WAIT_ASYNC();
    __syncthreads();
    for (int kt = 0; kt < nK; ++kt) {
        const int cur = kt & 1;
        if (kt + 1 < nK) stage(cur ^ 1, (kt + 1) * KT);  // overlap with WMMA
        compute(cur);
        WAIT_ASYNC();
        __syncthreads();   // next buffer ready AND current buffer free
    }
#else
    for (int kt = 0; kt < nK; ++kt) {
        const int k0 = kt * KT;
        uint4 av  = *(const uint4*)(Ap + k0);
        uint4 bv0 = *(const uint4*)(Bp + k0);
        uint4 bv1 = *(const uint4*)(Bp + k0 + 8);
        __syncthreads();
        *(uint4*)(&lA[0][ar * LSTR + ac])     = av;
        *(uint4*)(&lB[0][bn * LSTR + bc])     = bv0;
        *(uint4*)(&lB[0][bn * LSTR + bc + 8]) = bv1;
        __syncthreads();
        compute(0);
    }
#endif

    // epilogue: + bias, optional ReLU, store (B, E, N)
    const size_t orow0 = b0 + mrow + (size_t)hi * 8;   // VGPR r -> M = r + hi*8
#pragma unroll
    for (int j = 0; j < 4; ++j) {
        const int col = n0 + ncol + j * 16 + ln;
        const float bi = bias[(size_t)e * N + col];
#pragma unroll
        for (int r = 0; r < 8; ++r) {
            size_t base = ((orow0 + r) * E_ + e) * (size_t)N;
            float v = acc[j][r] + bi;
            if (RELU) v = fmaxf(v, 0.f);
            if (OUT_BF16) ((__bf16*)outv)[base + col] = (__bf16)v;
            else          ((float*)outv)[base + col]  = v;
        }
    }
}

// ---------------- Gram-Schmidt over experts + emb combine -------------------
__device__ __forceinline__ float block_reduce_sum(float v, float* red, int t) {
#pragma unroll
    for (int m = 16; m > 0; m >>= 1) v += __shfl_xor(v, m, 32);
    if ((t & 31) == 0) red[t >> 5] = v;
    __syncthreads();
    float s = red[0];
#pragma unroll
    for (int w = 1; w < 8; ++w) s += red[w];
    __syncthreads();   // red free for next reduction
    return s;
}

__global__ __launch_bounds__(256)
void gs_combine_kernel(const float* __restrict__ eo,
                       const float* __restrict__ emb,
                       __bf16* __restrict__ fo) {
    __shared__ float ex[E_][WID_];   // 32 KB
    __shared__ float red[8];

    const int b = blockIdx.x;
    const int t = threadIdx.x;
    const size_t base = (size_t)b * E_ * WID_;

    for (int e = 0; e < E_; ++e)
#pragma unroll
        for (int j = 0; j < 4; ++j)
            ex[e][t + 256 * j] = eo[base + (size_t)e * WID_ + t + 256 * j];
    __syncthreads();

    {   // row 0: normalize
        float p = 0.f;
#pragma unroll
        for (int j = 0; j < 4; ++j) { float v = ex[0][t + 256 * j]; p += v * v; }
        float inv = 1.f / sqrtf(block_reduce_sum(p, red, t));
#pragma unroll
        for (int j = 0; j < 4; ++j) ex[0][t + 256 * j] *= inv;
        __syncthreads();
    }

    for (int i = 1; i < E_; ++i) {   // classical GS like the reference
        float c[E_];
        for (int k = 0; k < i; ++k) {
            float p = 0.f;
#pragma unroll
            for (int j = 0; j < 4; ++j)
                p += ex[i][t + 256 * j] * ex[k][t + 256 * j];
            c[k] = block_reduce_sum(p, red, t);
        }
        float w[4], p = 0.f;
#pragma unroll
        for (int j = 0; j < 4; ++j) {
            float v = ex[i][t + 256 * j];
            for (int k = 0; k < i; ++k) v -= c[k] * ex[k][t + 256 * j];
            w[j] = v;
            p += v * v;
        }
        float inv = 1.f / sqrtf(block_reduce_sum(p, red, t));
        __syncthreads();
#pragma unroll
        for (int j = 0; j < 4; ++j) ex[i][t + 256 * j] = w[j] * inv;
        __syncthreads();
    }

    float ev[E_];
    for (int e = 0; e < E_; ++e) ev[e] = emb[(size_t)b * E_ + e];
#pragma unroll
    for (int j = 0; j < 4; ++j) {
        int k = t + 256 * j;
        float s = 0.f;
        for (int e = 0; e < E_; ++e) s += ex[e][k] * ev[e];
        fo[(size_t)b * WID_ + k] = (__bf16)fmaxf(s, 0.f);
    }
}

// ---------------- tid-bucketed WMMA head ------------------------------------
// grid (task, n-half, tile); block computes 64 gathered rows x 128 cols of
// out[row] = fo[row] @ W_head[task] + b_head[task] for rows in the task's
// bucket. Wht is [T][HD][WID]. Padding rows alias a valid row, never stored.
__global__ __launch_bounds__(256)
void head_wmma_kernel(const __bf16* __restrict__ fo,
                      const __bf16* __restrict__ Wht,
                      const float*  __restrict__ bh,
                      const int*    __restrict__ off,
                      const int*    __restrict__ perm,
                      float* __restrict__ out) {
    constexpr int K = WID_;
    __shared__ __align__(16) __bf16 lA[ASYNC_LDS ? 2 : 1][MT * LSTR];
    __shared__ __align__(16) __bf16 lB[ASYNC_LDS ? 2 : 1][NT * LSTR];
    __shared__ int lp[MT];

    const int task  = blockIdx.x;
    const int n0    = blockIdx.y * NT;          // 0 or 128 within HD_
    const int start = off[task] + blockIdx.z * MT;
    const int end   = off[task + 1];
    if (start >= end) return;   // uniform across block

    const int t = threadIdx.x;
    if (t < MT) {
        int r = start + t;
        lp[t] = perm[r < end ? r : end - 1];
    }
    __syncthreads();

    const int lane = t & 31;
    const int wave = t >> 5;
    const int mrow = (wave & 3) * 16;
    const int ncol = (wave >> 2) * 64;
    const int ln   = lane & 15;
    const int hi   = lane >> 4;

    v8f acc[4] = {};

    const int ar = t >> 2, ac = (t & 3) * 8;
    const int bn = t >> 1, bc = (t & 1) * 16;
    const __bf16* Ap = fo + (size_t)lp[ar] * K + ac;                       // gathered
    const __bf16* Bp = Wht + ((size_t)task * HD_ + n0 + bn) * (size_t)K + bc;

    auto compute = [&](int buf) {
        union F { uint4 q[2]; v16bf v; };
        F af;
        const __bf16* la = &lA[buf][(mrow + ln) * LSTR + hi * 8];
        af.q[0] = *(const uint4*)(la);
        af.q[1] = *(const uint4*)(la + 16);
#pragma unroll
        for (int j = 0; j < 4; ++j) {
            F bf;
            const __bf16* lb = &lB[buf][(ncol + j * 16 + ln) * LSTR + hi * 8];
            bf.q[0] = *(const uint4*)(lb);
            bf.q[1] = *(const uint4*)(lb + 16);
            acc[j] = __builtin_amdgcn_wmma_f32_16x16x32_bf16(
                false, af.v, false, bf.v, (short)0, acc[j], false, false);
        }
    };

    const int nK = K / KT;
#if ASYNC_LDS
    auto stage = [&](int buf, int k0) {
        async_copy_b128(Ap + k0, &lA[buf][ar * LSTR + ac]);
        async_copy_b128(Bp + k0,     &lB[buf][bn * LSTR + bc]);
        async_copy_b128(Bp + k0 + 8, &lB[buf][bn * LSTR + bc + 8]);
    };
    stage(0, 0);
    WAIT_ASYNC();
    __syncthreads();
    for (int kt = 0; kt < nK; ++kt) {
        const int cur = kt & 1;
        if (kt + 1 < nK) stage(cur ^ 1, (kt + 1) * KT);
        compute(cur);
        WAIT_ASYNC();
        __syncthreads();
    }
#else
    for (int kt = 0; kt < nK; ++kt) {
        const int k0 = kt * KT;
        uint4 av  = *(const uint4*)(Ap + k0);
        uint4 bv0 = *(const uint4*)(Bp + k0);
        uint4 bv1 = *(const uint4*)(Bp + k0 + 8);
        __syncthreads();
        *(uint4*)(&lA[0][ar * LSTR + ac])     = av;
        *(uint4*)(&lB[0][bn * LSTR + bc])     = bv0;
        *(uint4*)(&lB[0][bn * LSTR + bc + 8]) = bv1;
        __syncthreads();
        compute(0);
    }
#endif

    // scatter-store valid rows only
#pragma unroll
    for (int j = 0; j < 4; ++j) {
        const int col = n0 + ncol + j * 16 + ln;
        const float bi = bh[(size_t)task * HD_ + col];
#pragma unroll
        for (int r = 0; r < 8; ++r) {
            const int tr = mrow + hi * 8 + r;
            if (start + tr < end)
                out[(size_t)lp[tr] * HD_ + col] = acc[j][r] + bi;
        }
    }
}

// ---------------------------------------------------------------------------
static inline size_t align256(size_t x) { return (x + 255) & ~(size_t)255; }

extern "C" void kernel_launch(void* const* d_in, const int* in_sizes, int n_in,
                              void* d_out, int out_size, void* d_ws, size_t ws_size,
                              hipStream_t stream) {
    const float* x      = (const float*)d_in[0];
    const float* W_emb  = (const float*)d_in[1];
    const float* b_emb  = (const float*)d_in[2];
    const float* W0     = (const float*)d_in[3];
    const float* b0     = (const float*)d_in[4];
    const float* W1     = (const float*)d_in[5];
    const float* b1     = (const float*)d_in[6];
    const float* Wout   = (const float*)d_in[7];
    const float* bout   = (const float*)d_in[8];
    const float* W_head = (const float*)d_in[9];
    const float* b_head = (const float*)d_in[10];
    float* out = (float*)d_out;

    char* ws = (char*)d_ws;
    size_t off_b = 0;
    auto take = [&](size_t bytes) { size_t o = off_b; off_b = align256(off_b + bytes); return o; };

    const size_t nW0 = (size_t)E_ * FEAT_ * WID_;
    const size_t nW1 = (size_t)E_ * WID_ * WID_;
    const size_t nWo = nW1;
    const size_t nWh = (size_t)T_ * WID_ * HD_;

    __bf16* W0t  = (__bf16*)(ws + take(nW0 * 2));   // [E][N][K] bf16
    __bf16* W1t  = (__bf16*)(ws + take(nW1 * 2));
    __bf16* Wot  = (__bf16*)(ws + take(nWo * 2));
    __bf16* Wht  = (__bf16*)(ws + take(nWh * 2));   // [T][HD][WID] bf16
    __bf16* xb   = (__bf16*)(ws + take((size_t)B_ * FEAT_ * 2));
    __bf16* h0   = (__bf16*)(ws + take((size_t)B_ * E_ * WID_ * 2));
    __bf16* h1   = (__bf16*)(ws + take((size_t)B_ * E_ * WID_ * 2));
    float*  eo   = (float*) (ws + take((size_t)B_ * E_ * WID_ * 4));
    float*  emb  = (float*) (ws + take((size_t)B_ * E_ * 4));
    int*    tid  = (int*)   (ws + take((size_t)B_ * 4));
    __bf16* fo   = (__bf16*)(ws + take((size_t)B_ * WID_ * 2));
    int*    cnt  = (int*)   (ws + take((size_t)T_ * 4));
    int*    boff = (int*)   (ws + take((size_t)(T_ + 1) * 4));
    int*    cur  = (int*)   (ws + take((size_t)T_ * 4));
    int*    perm = (int*)   (ws + take((size_t)B_ * 4));
    (void)ws_size; (void)n_in; (void)in_sizes; (void)out_size;

    // 1) one-time conversion: transpose all weights to [.][N][K] bf16
    transpose_to_bf16_kernel<<<dim3(FEAT_ / 32, WID_ / 32, E_), 256, 0, stream>>>(W0,     W0t, FEAT_, WID_);
    transpose_to_bf16_kernel<<<dim3(WID_  / 32, WID_ / 32, E_), 256, 0, stream>>>(W1,     W1t, WID_,  WID_);
    transpose_to_bf16_kernel<<<dim3(WID_  / 32, WID_ / 32, E_), 256, 0, stream>>>(Wout,   Wot, WID_,  WID_);
    transpose_to_bf16_kernel<<<dim3(WID_  / 32, HD_  / 32, T_), 256, 0, stream>>>(W_head, Wht, WID_,  HD_);
    extract_x_kernel<<<(B_ * FEAT_) / 256, 256, 0, stream>>>(x, xb);
    emb_tid_kernel<<<B_ / 256, 256, 0, stream>>>(x, W_emb, b_emb, emb, tid);

    // 2) tid bucketing for the WMMA head (counters re-zeroed every call)
    bucket_init_kernel<<<1, 64, 0, stream>>>(cnt, cur);
    bucket_count_kernel<<<B_ / 256, 256, 0, stream>>>(tid, cnt);
    bucket_scan_kernel<<<1, 32, 0, stream>>>(cnt, boff);
    bucket_scatter_kernel<<<B_ / 256, 256, 0, stream>>>(tid, boff, cur, perm);

    // 3) three batched expert GEMMs on the WMMA units
    dim3 g(B_ / MT, WID_ / NT, E_), blk(256);
    gemm_moe_kernel<false, true, true ><<<g, blk, 0, stream>>>(xb, W0t, b0,   (void*)h0, FEAT_, WID_);
    gemm_moe_kernel<true,  true, true ><<<g, blk, 0, stream>>>(h0, W1t, b1,   (void*)h1, WID_,  WID_);
    gemm_moe_kernel<true,  false,false><<<g, blk, 0, stream>>>(h1, Wot, bout, (void*)eo, WID_,  WID_);

    // 4) Gram-Schmidt + expert combine (LDS-resident per batch row)
    gs_combine_kernel<<<B_, 256, 0, stream>>>(eo, emb, fo);

    // 5) gathered per-task head on WMMA (blocks beyond a bucket exit early)
    head_wmma_kernel<<<dim3(T_, HD_ / NT, B_ / MT), 256, 0, stream>>>(fo, Wht, b_head, boff, perm, out);
}
